// LNN_523986010666
// MI455X (gfx1250) — matrix operand order repaired and checked by reading, not compiled, not run on hardware
//
#include <hip/hip_runtime.h>
#include <hip/hip_bf16.h>

typedef __attribute__((ext_vector_type(16))) _Float16 v16h;
typedef __attribute__((ext_vector_type(8)))  _Float16 v8h;
typedef __attribute__((ext_vector_type(8)))  float    v8f;

#define WW 256   // hidden width
#define TS 16    // samples per workgroup

static __device__ inline v8f wmma16(v16h a, v16h b, v8f c) {
  // D = A(16x32 f16) * B(32x16 f16) + C(16x16 f32)
  return __builtin_amdgcn_wmma_f32_16x16x32_f16(
      /*neg_a=*/false, a, /*neg_b=*/false, b,
      /*c_mod=*/(short)0, c, /*reuse_a=*/false, /*reuse_b=*/false);
}

__global__ __launch_bounds__(256, 1) void lnn_hvp_kernel(
    const float* __restrict__ x,   // (B,2)
    const float* __restrict__ W1,  // (256,2)
    const float* __restrict__ b1,  // (256,)
    const float* __restrict__ W2,  // (256,256)
    const float* __restrict__ b2,  // (256,)
    const float* __restrict__ W3,  // (1,256)
    float* __restrict__ out,       // (B,2)
    int Bn)
{
  // ---- LDS (approx 302 KB of the 320 KB WGP pool) ----
  __shared__ __attribute__((aligned(32))) _Float16 W2s [WW*WW]; // row-major f16
  __shared__ __attribute__((aligned(32))) _Float16 W2Ts[WW*WW]; // transposed f16
  __shared__ __attribute__((aligned(32))) _Float16 h1T [TS*WW]; // [sample][i]
  __shared__ __attribute__((aligned(32))) _Float16 dh1T[TS*WW];
  __shared__ __attribute__((aligned(32))) _Float16 u2T [TS*WW];
  __shared__ __attribute__((aligned(32))) _Float16 du2T[TS*WW];
  __shared__ float w1c0[WW], w1c1[WW], b1v[WW], b2v[WW], w3v[WW];
  __shared__ float red[16][16];
  __shared__ float smax1[16], sS1[16], sP1[16];
  __shared__ float smax2[16], sS2[16], sP2[16], sQ2[16], sR2[16];
  __shared__ float sq1[16], sdq1[16], sdqB[16];
  __shared__ float sG0[16], sH0[16], sH1v[16];
  __shared__ float sz0[16], sz1[16];

  const int tid  = threadIdx.x;
  const int lane = tid & 31;
  const int wave = tid >> 5;
  const int hi   = (lane >> 4);    // lane half
  const int col  = lane & 15;

  // ================= Phase 0: stage inputs into LDS =================
  {
    // small vectors (blockDim == 256 == WW)
    w1c0[tid] = W1[tid * 2 + 0];
    w1c1[tid] = W1[tid * 2 + 1];   // == forward tangent of layer-1 preact (d = e1)
    b1v[tid]  = b1[tid];
    b2v[tid]  = b2[tid];
    w3v[tid]  = W3[tid];
    if (tid < TS) {
      int s = blockIdx.x * TS + tid;
      int sc = (s < Bn) ? s : (Bn - 1);
      sz0[tid] = x[2 * sc + 0];
      sz1[tid] = x[2 * sc + 1];
    }
    // W2 f32 -> f16, both row-major and transposed copies
    const int row = tid;
    const float4* src = reinterpret_cast<const float4*>(W2 + row * WW);
#pragma unroll 4
    for (int q = 0; q < WW / 4; ++q) {
      float4 v = src[q];
      int k = q * 4;
      _Float16 e0 = (_Float16)v.x, e1 = (_Float16)v.y;
      _Float16 e2 = (_Float16)v.z, e3 = (_Float16)v.w;
      W2s[row * WW + k + 0] = e0;  W2s[row * WW + k + 1] = e1;
      W2s[row * WW + k + 2] = e2;  W2s[row * WW + k + 3] = e3;
      W2Ts[(k + 0) * WW + row] = e0;  W2Ts[(k + 1) * WW + row] = e1;
      W2Ts[(k + 2) * WW + row] = e2;  W2Ts[(k + 3) * WW + row] = e3;
    }
  }
  __syncthreads();

  // ============ Phase 1: layer-1 softmax h1 and tangent dh1 ============
  // thread t handles sample j = t&15, chunk c = t>>4 -> rows [16c,16c+16)
  {
    const int j = tid & 15, c = tid >> 4, i0 = c * 16;
    const float z0 = sz0[j], z1 = sz1[j];
    float m = -3.4e38f;
#pragma unroll
    for (int i = i0; i < i0 + 16; ++i) {
      float a = fmaf(w1c0[i], z0, fmaf(w1c1[i], z1, b1v[i]));
      m = fmaxf(m, a);
    }
    red[c][j] = m;
    __syncthreads();
    if (tid < TS) {
      float mm = -3.4e38f;
#pragma unroll
      for (int cc = 0; cc < 16; ++cc) mm = fmaxf(mm, red[cc][tid]);
      smax1[tid] = mm; sS1[tid] = 0.f; sP1[tid] = 0.f;
    }
    __syncthreads();
    const float mj = smax1[j];
    float S = 0.f, P = 0.f;
#pragma unroll
    for (int i = i0; i < i0 + 16; ++i) {
      float a = fmaf(w1c0[i], z0, fmaf(w1c1[i], z1, b1v[i]));
      float e = __expf(a - mj);
      S += e; P += e * w1c1[i];
    }
    atomicAdd(&sS1[j], S);   // ds_add_f32
    atomicAdd(&sP1[j], P);
    __syncthreads();
    const float invS = 1.f / sS1[j];
    const float t1   = sP1[j] * invS;          // h1 . da1
#pragma unroll
    for (int i = i0; i < i0 + 16; ++i) {
      float a = fmaf(w1c0[i], z0, fmaf(w1c1[i], z1, b1v[i]));
      float h = __expf(a - mj) * invS;
      h1T [j * WW + i] = (_Float16)h;
      dh1T[j * WW + i] = (_Float16)(h * (w1c1[i] - t1));  // J_s1 * da1
    }
  }
  __syncthreads();

  // ============ Phase 2: GEMM1  a2 = W2*h1, da2 = W2*dh1 (registers) ====
  v8f acc[2][2];   // [m-tile][0:a2, 1:da2], rows = W2 rows, cols = samples
#pragma unroll
  for (int t = 0; t < 2; ++t) { acc[t][0] = (v8f)0.f; acc[t][1] = (v8f)0.f; }
#pragma unroll
  for (int t = 0; t < 2; ++t) {
    const int m = wave * 2 + t;
    const int arow = m * 16 + col;             // A rows 0..15 in both halves
#pragma unroll
    for (int k = 0; k < 8; ++k) {
      const int K0 = k * 32 + hi * 16;
      v8h lo = *(const v8h*)&W2s[arow * WW + K0];
      v8h hp = *(const v8h*)&W2s[arow * WW + K0 + 8];
      v16h a;
#pragma unroll
      for (int e = 0; e < 8; ++e) { a[e] = lo[e]; a[e + 8] = hp[e]; }
      v16h bh  = *(const v16h*)&h1T [col * WW + k * 32 + hi * 16];
      v16h bdh = *(const v16h*)&dh1T[col * WW + k * 32 + hi * 16];
      acc[t][0] = wmma16(a, bh,  acc[t][0]);
      acc[t][1] = wmma16(a, bdh, acc[t][1]);
    }
  }

  // ============ Phase 3: layer-2 softmax/tangent/cotangent from regs ====
  // lane element (t,r): row = (2*wave+t)*16 + 8*hi + r, column sample = col
  float a2l[16], da2l[16];
#pragma unroll
  for (int t = 0; t < 2; ++t)
#pragma unroll
    for (int r = 0; r < 8; ++r) {
      int row = (wave * 2 + t) * 16 + hi * 8 + r;
      a2l [t * 8 + r] = acc[t][0][r] + b2v[row];
      da2l[t * 8 + r] = acc[t][1][r];
    }
  {
    float lmax = -3.4e38f;
#pragma unroll
    for (int e = 0; e < 16; ++e) lmax = fmaxf(lmax, a2l[e]);
    red[wave * 2 + hi][col] = lmax;            // 16 unique partials per column
  }
  __syncthreads();
  if (tid < TS) {
    float mm = -3.4e38f;
#pragma unroll
    for (int p = 0; p < 16; ++p) mm = fmaxf(mm, red[p][tid]);
    smax2[tid] = mm;
    sS2[tid] = 0.f; sP2[tid] = 0.f; sQ2[tid] = 0.f; sR2[tid] = 0.f;
  }
  __syncthreads();
  {
    const float m2 = smax2[col];
    float S = 0.f, P = 0.f, Q = 0.f, R = 0.f;
#pragma unroll
    for (int t = 0; t < 2; ++t)
#pragma unroll
      for (int r = 0; r < 8; ++r) {
        int row = (wave * 2 + t) * 16 + hi * 8 + r;
        float e  = __expf(a2l[t * 8 + r] - m2);
        float da = da2l[t * 8 + r], w = w3v[row];
        S += e; P += e * da; Q += e * w; R += e * w * da;
      }
    atomicAdd(&sS2[col], S); atomicAdd(&sP2[col], P);
    atomicAdd(&sQ2[col], Q); atomicAdd(&sR2[col], R);
  }
  __syncthreads();
  if (tid < TS) {
    float S = sS2[tid], iS = 1.f / S;
    float t2 = sP2[tid] * iS, s2 = sQ2[tid] * iS;
    float ds2 = sR2[tid] * iS - s2 * t2;
    sP2[tid] = t2; sQ2[tid] = s2; sR2[tid] = ds2; sS2[tid] = iS;
    sq1[tid] = 0.f; sdq1[tid] = 0.f; sdqB[tid] = 0.f;  // for phase 5
  }
  __syncthreads();
  {
    const float m2 = smax2[col], iS = sS2[col];
    const float t2 = sP2[col], s2 = sQ2[col], ds2 = sR2[col];
#pragma unroll
    for (int t = 0; t < 2; ++t) {
      v8h ub, dub;
#pragma unroll
      for (int r = 0; r < 8; ++r) {
        int row = (wave * 2 + t) * 16 + hi * 8 + r;
        float h2  = __expf(a2l[t * 8 + r] - m2) * iS;
        float dh2 = h2 * (da2l[t * 8 + r] - t2);
        float wms = w3v[row] - s2;
        ub [r] = (_Float16)(h2 * wms);               // u2  = J_s2 w3
        dub[r] = (_Float16)(dh2 * wms - h2 * ds2);   // du2 = tangent of u2
      }
      int M0 = (wave * 2 + t) * 16 + hi * 8;
      *(v8h*)&u2T [col * WW + M0] = ub;    // transposed: [sample][i]
      *(v8h*)&du2T[col * WW + M0] = dub;
    }
  }
  __syncthreads();

  // ===== Phase 4: GEMM2  u1 = W2^T u2, du1 = W2^T du2  (as u2^T * W2) ====
  v8f accu[2][2];  // [n-tile][0:u1, 1:du1], rows = samples, cols = W2 columns
#pragma unroll
  for (int t = 0; t < 2; ++t) { accu[t][0] = (v8f)0.f; accu[t][1] = (v8f)0.f; }
#pragma unroll
  for (int t = 0; t < 2; ++t) {
    const int n = wave * 2 + t;
    const int colN = n * 16 + col;             // B column = W2 column index
#pragma unroll
    for (int k = 0; k < 8; ++k) {
      const int K0 = k * 32 + hi * 16;
      v8h lo0 = *(const v8h*)&u2T [col * WW + K0];
      v8h hp0 = *(const v8h*)&u2T [col * WW + K0 + 8];
      v8h lo1 = *(const v8h*)&du2T[col * WW + K0];
      v8h hp1 = *(const v8h*)&du2T[col * WW + K0 + 8];
      v16h au, adu;
#pragma unroll
      for (int e = 0; e < 8; ++e) {
        au [e] = lo0[e]; au [e + 8] = hp0[e];
        adu[e] = lo1[e]; adu[e + 8] = hp1[e];
      }
      v16h bw = *(const v16h*)&W2Ts[colN * WW + k * 32 + hi * 16];
      accu[t][0] = wmma16(au,  bw, accu[t][0]);
      accu[t][1] = wmma16(adu, bw, accu[t][1]);
    }
  }

  // ============ Phase 5: epilogue reductions and output ================
  // lane element (t,r): sample j = 8*hi + r, feature ip = (2*wave+t)*16 + col
  float hv[16], dhv[16];
#pragma unroll
  for (int t = 0; t < 2; ++t)
#pragma unroll
    for (int r = 0; r < 8; ++r) {
      int j  = hi * 8 + r;
      int ip = (wave * 2 + t) * 16 + col;
      hv [t * 8 + r] = (float)h1T [j * WW + ip];
      dhv[t * 8 + r] = (float)dh1T[j * WW + ip];
      float u1v  = accu[t][0][r];
      float du1v = accu[t][1][r];
      atomicAdd(&sq1 [j], hv [t * 8 + r] * u1v);   // q1  = h1 . u1
      atomicAdd(&sdq1[j], dhv[t * 8 + r] * u1v);   // dq1 part A = dh1 . u1
      atomicAdd(&sdqB[j], hv [t * 8 + r] * du1v);  // dq1 part B = h1 . du1
    }
  __syncthreads();
  if (tid < TS) {
    sdq1[tid] += sdqB[tid];
    sG0[tid] = 0.f; sH0[tid] = 0.f; sH1v[tid] = 0.f;
  }
  __syncthreads();
#pragma unroll
  for (int t = 0; t < 2; ++t)
#pragma unroll
    for (int r = 0; r < 8; ++r) {
      int j  = hi * 8 + r;
      int ip = (wave * 2 + t) * 16 + col;
      float h  = hv [t * 8 + r], dh = dhv[t * 8 + r];
      float u1v = accu[t][0][r], du1v = accu[t][1][r];
      float q1 = sq1[j], dq1 = sdq1[j];
      float v1  = h * u1v - h * q1;                          // J_s1 u1
      float dv1 = dh * u1v + h * du1v - dh * q1 - h * dq1;   // tangent
      atomicAdd(&sG0 [j], w1c0[ip] * v1);    // g[0]   = W1[:,0] . v1
      atomicAdd(&sH0 [j], w1c0[ip] * dv1);   // H[0,1] = W1[:,0] . dv1
      atomicAdd(&sH1v[j], w1c1[ip] * dv1);   // H[1,1] = W1[:,1] . dv1
    }
  __syncthreads();
  if (tid < TS) {
    int s = blockIdx.x * TS + tid;
    if (s < Bn) {
      float z1 = sz1[tid];
      out[2 * s + 0] = z1;
      out[2 * s + 1] = (sG0[tid] - sH0[tid] * z1) / sH1v[tid];
    }
  }
}

extern "C" void kernel_launch(void* const* d_in, const int* in_sizes, int n_in,
                              void* d_out, int out_size, void* d_ws, size_t ws_size,
                              hipStream_t stream) {
  const float* x  = (const float*)d_in[0];
  const float* W1 = (const float*)d_in[1];
  const float* b1 = (const float*)d_in[2];
  const float* W2 = (const float*)d_in[3];
  const float* b2 = (const float*)d_in[4];
  const float* W3 = (const float*)d_in[5];
  // d_in[6] = b3 : constant offset, cancels in grad/hessian — unused.
  float* out = (float*)d_out;

  int Bn = in_sizes[0] / 2;          // samples (x is B x 2)
  int grid = (Bn + TS - 1) / TS;
  lnn_hvp_kernel<<<grid, 256, 0, stream>>>(x, W1, b1, W2, b2, W3, out, Bn);
}
